// MambaSSMTest_79182017069464
// MI455X (gfx1250) — matrix operand order repaired and checked by reading, not compiled, not run on hardware
//
#include <hip/hip_runtime.h>
#include <stdint.h>

// ---------------------------------------------------------------------------
// MambaSSM fused pipeline for gfx1250 (MI455X, wave32, WMMA)
//   K1: xdb  = x @ W_xproj^T   (8192 x 96,  K=1024)  bf16 WMMA, 6 acc/wave
//   K2: dt   = softplus(xdb[:, :64] @ W_dt^T + b_dt) bf16 WMMA, 4 acc/wave
//   K3: fused selective scan over L, h[16] in registers, pooled mean
//   K4: logits = pooled @ W_cls^T + b_cls            bf16 WMMA
// Fragments staged in LDS fragment-major (2x ds_load_b128 per operand);
// bf16 conversion uses v_cvt_pk_bf16_f32 when the builtin is available.
// ---------------------------------------------------------------------------

typedef __attribute__((ext_vector_type(16))) __bf16    v16bf;
typedef __attribute__((ext_vector_type(2)))  __bf16    v2bf;
typedef __attribute__((ext_vector_type(8)))  uint32_t  u32x8;
typedef __attribute__((ext_vector_type(8)))  float     v8f;

#define BM    8192   // B*L rows
#define DD    1024   // model dim
#define KPROJ 96     // R + 2N
#define RR    64
#define NN    16
#define NCLS  1000
#define LL    1024
#define BB    8

// packed bf16 convert: one v_cvt_pk_bf16_f32 when available
__device__ __forceinline__ uint32_t pack2bf(float lo, float hi) {
#if __has_builtin(__builtin_amdgcn_cvt_pk_bf16_f32)
  v2bf p = __builtin_amdgcn_cvt_pk_bf16_f32(lo, hi);
  return __builtin_bit_cast(uint32_t, p);
#else
  __bf16 l = (__bf16)lo, h = (__bf16)hi;
  return (uint32_t)__builtin_bit_cast(unsigned short, l) |
         ((uint32_t)__builtin_bit_cast(unsigned short, h) << 16);
#endif
}
// one 32-byte vector load from LDS -> 2x ds_load_b128
__device__ __forceinline__ v16bf loadFrag(const uint32_t* p) {
  u32x8 v = *(const u32x8*)p;
  return __builtin_bit_cast(v16bf, v);
}

// Inverse WMMA layout maps (even k, pair {k,k+1} packs into one dword):
// A 16x32 (MxK): lane = ((k>>3)&1)*16 + m ; dword = (k>>4)*4 + ((k&7)>>1)
__device__ __forceinline__ void mapA(int m, int k, int& lane, int& dw) {
  lane = (((k >> 3) & 1) << 4) + m;
  dw   = ((k >> 4) << 2) + ((k & 7) >> 1);
}
// B 32x16 (KxN): lane = ((k>>3)&1)*16 + n ; dword = (k>>4)*4 + ((k&7)>>1)
__device__ __forceinline__ void mapB(int n, int k, int& lane, int& dw) {
  lane = (((k >> 3) & 1) << 4) + n;
  dw   = ((k >> 4) << 2) + ((k & 7) >> 1);
}

// ------------------------- K1: x @ W_xproj^T -------------------------------
// block = 128 (4 waves). Block tile: M=64 (wave-per-16-rows) x N=96 (6 accs).
__global__ void k_xproj(const float* __restrict__ x, const float* __restrict__ W,
                        float* __restrict__ xdb) {
  __shared__ __align__(32) uint32_t fragA[4][32][8];
  __shared__ __align__(32) uint32_t fragB[6][32][8];
  const int tid = threadIdx.x, lane = tid & 31, wave = tid >> 5;
  const int m0 = blockIdx.x * 64;
  v8f acc[6] = {};
  for (int k0 = 0; k0 < DD; k0 += 32) {
    // stage B: 6 tiles x 16 rows x 16 pairs = 1536 dwords, 12 per thread
#pragma unroll
    for (int it = 0; it < 12; ++it) {
      int e = it * 128 + tid;
      int t = e >> 8, e2 = e & 255;
      int n = e2 >> 4, k = (e2 & 15) << 1;
      const float* src = &W[(size_t)(t * 16 + n) * DD + k0 + k];
      int bl, dw; mapB(n, k, bl, dw);
      fragB[t][bl][dw] = pack2bf(src[0], src[1]);
    }
    // stage A: 4 waves x 256 pairs = 1024 dwords, 8 per thread
#pragma unroll
    for (int it = 0; it < 8; ++it) {
      int e = it * 128 + tid;
      int w = e >> 8, e2 = e & 255;
      int m = e2 >> 4, k = (e2 & 15) << 1;
      const float* src = &x[(size_t)(m0 + w * 16 + m) * DD + k0 + k];
      int al, dw; mapA(m, k, al, dw);
      fragA[w][al][dw] = pack2bf(src[0], src[1]);
    }
    __syncthreads();
    v16bf a = loadFrag(&fragA[wave][lane][0]);
#pragma unroll
    for (int t = 0; t < 6; ++t) {
      v16bf b = loadFrag(&fragB[t][lane][0]);
      acc[t] = __builtin_amdgcn_wmma_f32_16x16x32_bf16(false, a, false, b,
                                                       (short)0, acc[t], false, false);
    }
    __syncthreads();
  }
  const int n = lane & 15, half = lane >> 4;
#pragma unroll
  for (int t = 0; t < 6; ++t)
#pragma unroll
    for (int r = 0; r < 8; ++r) {
      int m = m0 + wave * 16 + half * 8 + r;
      xdb[(size_t)m * KPROJ + t * 16 + n] = acc[t][r];
    }
}

// -------------------- K2: softplus(dt_r @ W_dt^T + b) ----------------------
// block = 128 (4 waves). Block tile: M=64 x N=64 (4 accs/wave). K=64.
__global__ void k_dt(const float* __restrict__ xdb, const float* __restrict__ Wdt,
                     const float* __restrict__ bdt, float* __restrict__ dtbuf) {
  __shared__ __align__(32) uint32_t fragA[4][32][8];
  __shared__ __align__(32) uint32_t fragB[4][32][8];
  const int tid = threadIdx.x, lane = tid & 31, wave = tid >> 5;
  const int m0 = blockIdx.x * 64;
  const int n0 = blockIdx.y * 64;
  v8f acc[4] = {};
  for (int k0 = 0; k0 < RR; k0 += 32) {
#pragma unroll
    for (int it = 0; it < 8; ++it) {           // B: 4 tiles x 256 = 1024 dwords
      int e = it * 128 + tid;
      int t = e >> 8, e2 = e & 255;
      int n = e2 >> 4, k = (e2 & 15) << 1;
      const float* src = &Wdt[(size_t)(n0 + t * 16 + n) * RR + k0 + k];
      int bl, dw; mapB(n, k, bl, dw);
      fragB[t][bl][dw] = pack2bf(src[0], src[1]);
    }
#pragma unroll
    for (int it = 0; it < 8; ++it) {           // A: 4 waves x 256 = 1024 dwords
      int e = it * 128 + tid;
      int w = e >> 8, e2 = e & 255;
      int m = e2 >> 4, k = (e2 & 15) << 1;
      const float* src = &xdb[(size_t)(m0 + w * 16 + m) * KPROJ + k0 + k];
      int al, dw; mapA(m, k, al, dw);
      fragA[w][al][dw] = pack2bf(src[0], src[1]);
    }
    __syncthreads();
    v16bf a = loadFrag(&fragA[wave][lane][0]);
#pragma unroll
    for (int t = 0; t < 4; ++t) {
      v16bf b = loadFrag(&fragB[t][lane][0]);
      acc[t] = __builtin_amdgcn_wmma_f32_16x16x32_bf16(false, a, false, b,
                                                       (short)0, acc[t], false, false);
    }
    __syncthreads();
  }
  const int n = lane & 15, half = lane >> 4;
#pragma unroll
  for (int t = 0; t < 4; ++t)
#pragma unroll
    for (int r = 0; r < 8; ++r) {
      int m = m0 + wave * 16 + half * 8 + r;
      int d = n0 + t * 16 + n;
      float z = acc[t][r] + bdt[d];
      float sp = (z > 20.f) ? z : __logf(1.f + __expf(z));   // softplus
      dtbuf[(size_t)m * DD + d] = sp;
    }
}

// ------------------ K3: fused selective scan + mean pool -------------------
// One thread per (b, d). h[16] in registers; Bm/Cm staged in LDS per 16-step
// chunk (shared by all 256 d-threads of the same b).
__global__ void k_scan(const float* __restrict__ x, const float* __restrict__ A_log,
                       const float* __restrict__ Dp, const float* __restrict__ xdb,
                       const float* __restrict__ dtbuf, float* __restrict__ pooled) {
  const int tid = threadIdx.x;
  const int d = blockIdx.x * 256 + tid;
  const int b = blockIdx.y;

  float A[NN];
#pragma unroll
  for (int n = 0; n < NN; ++n) A[n] = -__expf(A_log[d * NN + n]);
  float h[NN];
#pragma unroll
  for (int n = 0; n < NN; ++n) h[n] = 0.f;
  const float dparam = Dp[d];
  float acc = 0.f;

  __shared__ float sB[16][NN];
  __shared__ float sC[16][NN];
  const size_t rowBase = (size_t)b * LL;

  for (int c = 0; c < LL / 16; ++c) {
    __syncthreads();
#pragma unroll
    for (int it = 0; it < 2; ++it) {          // 512 floats staged by 256 threads
      int e = it * 256 + tid;
      int lo = e >> 5, col = e & 31;          // 16 l-steps x 32 (Bm|Cm)
      float v = xdb[(rowBase + c * 16 + lo) * KPROJ + RR + col];
      if (col < NN) sB[lo][col] = v; else sC[lo][col - NN] = v;
    }
    __syncthreads();
#pragma unroll 4
    for (int lo = 0; lo < 16; ++lo) {
      size_t row = rowBase + c * 16 + lo;
      float dtv = dtbuf[row * DD + d];
      float xv  = x[row * DD + d];
      float y = 0.f;
#pragma unroll
      for (int n = 0; n < NN; ++n) {
        float dA = __expf(dtv * A[n]);
        h[n] = dA * h[n] + dtv * sB[lo][n] * xv;   // h = dA*h + dt*B*x
        y += h[n] * sC[lo][n];                      // y = <h, C>
      }
      acc += y + dparam * xv;
    }
    if (c + 1 < LL / 16) {                    // global_prefetch_b8 for next chunk
      __builtin_prefetch(&dtbuf[(rowBase + (c + 1) * 16) * DD + d], 0, 1);
      __builtin_prefetch(&x[(rowBase + (c + 1) * 16) * DD + d], 0, 1);
    }
  }
  pooled[(size_t)b * DD + d] = acc * (1.f / (float)LL);
}

// ---------------------- K4: classifier head (WMMA) -------------------------
// One wave per 16-class tile; M=16 (rows 8..15 zero), K=1024.
__global__ void k_cls(const float* __restrict__ pooled, const float* __restrict__ Wc,
                      const float* __restrict__ bc, float* __restrict__ out) {
  __shared__ __align__(32) uint32_t fragA[32][8];
  __shared__ __align__(32) uint32_t fragB[32][8];
  const int lane = threadIdx.x;
  const int n0 = blockIdx.x * 16;
  v8f acc = {};
  for (int k0 = 0; k0 < DD; k0 += 32) {
#pragma unroll
    for (int it = 0; it < 8; ++it) {          // 256 pairs each for A and B
      int e = it * 32 + lane;
      int row = e >> 4, k = (e & 15) << 1;
      float a0 = 0.f, a1 = 0.f;
      if (row < BB) {
        const float* s = &pooled[(size_t)row * DD + k0 + k];
        a0 = s[0]; a1 = s[1];
      }
      int al, adw; mapA(row, k, al, adw);
      fragA[al][adw] = pack2bf(a0, a1);
      float b0 = 0.f, b1 = 0.f;
      int cls = n0 + row;
      if (cls < NCLS) {
        const float* s = &Wc[(size_t)cls * DD + k0 + k];
        b0 = s[0]; b1 = s[1];
      }
      int bl, bdw; mapB(row, k, bl, bdw);
      fragB[bl][bdw] = pack2bf(b0, b1);
    }
    __syncthreads();
    v16bf a = loadFrag(&fragA[lane][0]);
    v16bf b = loadFrag(&fragB[lane][0]);
    acc = __builtin_amdgcn_wmma_f32_16x16x32_bf16(false, a, false, b,
                                                  (short)0, acc, false, false);
    __syncthreads();
  }
  const int n = lane & 15, half = lane >> 4;
  if (half == 0) {                            // rows 0..7 are the valid batch
#pragma unroll
    for (int r = 0; r < 8; ++r) {
      int c = n0 + n;
      if (c < NCLS) out[(size_t)r * NCLS + c] = acc[r] + bc[c];
    }
  }
}

// ---------------------------------------------------------------------------
extern "C" void kernel_launch(void* const* d_in, const int* in_sizes, int n_in,
                              void* d_out, int out_size, void* d_ws, size_t ws_size,
                              hipStream_t stream) {
  (void)in_sizes; (void)n_in; (void)out_size; (void)ws_size;
  const float* x       = (const float*)d_in[0];
  const float* A_log   = (const float*)d_in[1];
  const float* D_param = (const float*)d_in[2];
  const float* W_xproj = (const float*)d_in[3];
  const float* W_dt    = (const float*)d_in[4];
  const float* b_dt    = (const float*)d_in[5];
  const float* W_cls   = (const float*)d_in[6];
  const float* b_cls   = (const float*)d_in[7];
  float* out = (float*)d_out;

  char* ws = (char*)d_ws;
  float* xdb    = (float*)(ws);                                   //  3,145,728 B
  float* dtbuf  = (float*)(ws + (size_t)BM * KPROJ * 4);          // 33,554,432 B
  float* pooled = (float*)(ws + (size_t)BM * KPROJ * 4
                              + (size_t)BM * DD * 4);             //     32,768 B

  k_xproj<<<dim3(BM / 64),            128, 0, stream>>>(x, W_xproj, xdb);
  k_dt   <<<dim3(BM / 64, DD / 64),   128, 0, stream>>>(xdb, W_dt, b_dt, dtbuf);
  k_scan <<<dim3(DD / 256, BB),       256, 0, stream>>>(x, A_log, D_param, xdb,
                                                        dtbuf, pooled);
  k_cls  <<<(NCLS + 15) / 16,          32, 0, stream>>>(pooled, W_cls, b_cls, out);
}
